// MShuffle_42932493091208
// MI455X (gfx1250) — compile-verified, matrix operand order checked by reading, not run on hardware
//
#include <hip/hip_runtime.h>

#define NDIM  4096
#define HDIM  2048
#define DEPTH 12
#define BATCH 4096

#define BM 256            // batch rows per block
#define BN 128            // output cols per block
#define BK 32             // K per chunk (one bf16 WMMA K-step)
#define LDR 40            // LDS row stride in bf16 elems (80 B, multiple of 16 B)
#define NKCHUNK (NDIM / BK)

typedef __attribute__((ext_vector_type(16))) __bf16 v16bf;
typedef __attribute__((ext_vector_type(8)))  __bf16 v8bf;
typedef __attribute__((ext_vector_type(4)))  __bf16 v4bf;
typedef __attribute__((ext_vector_type(8)))  float  v8f;
typedef __attribute__((ext_vector_type(4)))  int    v4i;

typedef __attribute__((address_space(1))) v4i gas_v4i;   // global int4
typedef __attribute__((address_space(3))) v4i las_v4i;   // LDS int4

#if __has_builtin(__builtin_amdgcn_global_load_async_to_lds_b128)
#define HAVE_ASYNC_LDS 1
#else
#define HAVE_ASYNC_LDS 0
#endif

// 16-byte global -> LDS copy: async (ASYNCcnt, no VGPR dest) if available.
__device__ __forceinline__ void cp16_g2l(const void* g, void* l) {
#if HAVE_ASYNC_LDS
    __builtin_amdgcn_global_load_async_to_lds_b128(
        (gas_v4i*)g, (las_v4i*)l, 0, 0);
#else
    *(v8bf*)l = *(const v8bf*)g;
#endif
}

__device__ __forceinline__ void wait_stage() {
#if HAVE_ASYNC_LDS
# if __has_builtin(__builtin_amdgcn_s_wait_asynccnt)
    __builtin_amdgcn_s_wait_asynccnt(0);
# else
    asm volatile("s_wait_asynccnt 0x0" ::: "memory");
# endif
#endif
}

// ---------------------------------------------------------------------------
// h = softplus(xbf @ wbf^T + b), all-bf16 operands, f32 accumulate.
// 256 threads = 8 waves; block tile 256x128; wave tile 64x64 (4x4 subtiles).
// Double-buffered LDS staging via async global->LDS copies.
// ---------------------------------------------------------------------------
__global__ void __launch_bounds__(256)
mshuffle_gemm_softplus(const __bf16* __restrict__ xbf,   // [BATCH, NDIM]
                       const __bf16* __restrict__ wbf,   // [HDIM, NDIM]
                       const float*  __restrict__ bias,  // [HDIM]
                       float*        __restrict__ h)     // [BATCH, HDIM]
{
    __shared__ __align__(16) __bf16 As[2][BM * LDR];
    __shared__ __align__(16) __bf16 Bs[2][BN * LDR];

    const int row0 = blockIdx.x * BM;
    const int col0 = blockIdx.y * BN;
    const int t    = threadIdx.x;
    const int lane = t & 31;
    const int w    = t >> 5;     // 0..7
    const int wr   = w & 3;      // m group: subtiles 4*wr .. 4*wr+3
    const int wc   = w >> 2;     // n group: subtiles 4*wc .. 4*wc+3

    const __bf16* Ag = xbf + (size_t)row0 * NDIM;
    const __bf16* Bg = wbf + (size_t)col0 * NDIM;

    v8f acc[4][4] = {};

    // CDNA5 16-bit A/B fragment layout: lane L holds row/col (L&15),
    // K-base (L<16?0:8), two contiguous 8-elem (16 B) runs at K +0 and +16.
    const int mloc = lane & 15;
    const int kb   = (lane < 16) ? 0 : 8;

    auto stage = [&](int buf, int k0) {
        #pragma unroll
        for (int i = 0; i < 4; ++i) {             // A: 256 rows x 4 segs = 1024
            const int seg = t + i * 256;
            const int r = seg >> 2, s = (seg & 3) * 8;   // s in bf16 elems
            cp16_g2l(Ag + (size_t)r * NDIM + k0 + s, &As[buf][r * LDR + s]);
        }
        #pragma unroll
        for (int i = 0; i < 2; ++i) {             // B: 128 rows x 4 segs = 512
            const int seg = t + i * 256;
            const int r = seg >> 2, s = (seg & 3) * 8;
            cp16_g2l(Bg + (size_t)r * NDIM + k0 + s, &Bs[buf][r * LDR + s]);
        }
    };

    stage(0, 0);
    wait_stage();
    __syncthreads();

    for (int kc = 0; kc < NKCHUNK; ++kc) {
        const int cur = kc & 1;
        if (kc + 1 < NKCHUNK) stage(cur ^ 1, (kc + 1) * BK);   // overlaps compute

        v16bf bfrag[4];
        #pragma unroll
        for (int nj = 0; nj < 4; ++nj) {
            const __bf16* p = &Bs[cur][((wc * 4 + nj) * 16 + mloc) * LDR + kb];
            v8bf lo = *(const v8bf*)p;            // K = kb .. kb+7
            v8bf hi = *(const v8bf*)(p + 16);     // K = kb+16 .. kb+23
            bfrag[nj] = __builtin_shufflevector(lo, hi,
                0,1,2,3,4,5,6,7,8,9,10,11,12,13,14,15);
        }
        #pragma unroll
        for (int mi = 0; mi < 4; ++mi) {
            const __bf16* p = &As[cur][((wr * 4 + mi) * 16 + mloc) * LDR + kb];
            v8bf lo = *(const v8bf*)p;
            v8bf hi = *(const v8bf*)(p + 16);
            v16bf afrag = __builtin_shufflevector(lo, hi,
                0,1,2,3,4,5,6,7,8,9,10,11,12,13,14,15);
            #pragma unroll
            for (int nj = 0; nj < 4; ++nj) {
                acc[mi][nj] = __builtin_amdgcn_wmma_f32_16x16x32_bf16(
                    false, afrag, false, bfrag[nj],
                    (short)0, acc[mi][nj], false, false);
            }
        }

        wait_stage();        // next chunk fully in LDS
        __syncthreads();     // all waves done with 'cur' before it is re-staged
    }

    // Epilogue: bias + softplus. C/D layout: VGPR r, lane L ->
    // row = r + 8*(L>>4), col = L&15 within the 16x16 subtile.
    const int nloc = lane & 15;
    const int rh   = (lane >> 4) * 8;
    #pragma unroll
    for (int mi = 0; mi < 4; ++mi) {
        #pragma unroll
        for (int nj = 0; nj < 4; ++nj) {
            const int col = col0 + (wc * 4 + nj) * 16 + nloc;
            const float bv = bias[col];
            #pragma unroll
            for (int r = 0; r < 8; ++r) {
                const int row = row0 + (wr * 4 + mi) * 16 + r + rh;
                const float v  = acc[mi][nj][r] + bv;
                const float sp = (v > 20.0f) ? v : __logf(1.0f + __expf(v));
                h[(size_t)row * HDIM + col] = sp;
            }
        }
    }
}

// ---------------------------------------------------------------------------
// Scatter-add: xw[b, idx[j]] += h[b, j]; keep bf16 mirror coherent.
// idx injective per step -> no atomics needed.
// ---------------------------------------------------------------------------
__global__ void __launch_bounds__(256)
mshuffle_scatter(float* __restrict__ xw, __bf16* __restrict__ xbf,
                 const float* __restrict__ h, const int* __restrict__ idx)
{
    const size_t total  = (size_t)BATCH * HDIM;
    const size_t stride = (size_t)gridDim.x * blockDim.x;
    for (size_t i = (size_t)blockIdx.x * blockDim.x + threadIdx.x; i < total; i += stride) {
        const size_t b = i >> 11;            // / HDIM
        const int    j = (int)(i & (HDIM - 1));
        const size_t o = b * NDIM + idx[j];
        const float  v = xw[o] + h[i];
        xw[o]  = v;
        xbf[o] = (__bf16)v;
    }
}

// ---------------------------------------------------------------------------
// Init: xw = x (f32) and xbf = bf16(x)
// ---------------------------------------------------------------------------
__global__ void __launch_bounds__(256)
mshuffle_init_x(float4* __restrict__ xw, v4bf* __restrict__ xbf,
                const float4* __restrict__ x, int n4)
{
    const int stride = gridDim.x * blockDim.x;
    for (int i = blockIdx.x * blockDim.x + threadIdx.x; i < n4; i += stride) {
        const float4 v = x[i];
        xw[i] = v;
        v4bf o = { (__bf16)v.x, (__bf16)v.y, (__bf16)v.z, (__bf16)v.w };
        xbf[i] = o;
    }
}

// ---------------------------------------------------------------------------
// Per-step weight conversion: wbf = bf16(W_d)
// ---------------------------------------------------------------------------
__global__ void __launch_bounds__(256)
mshuffle_convert_w(v4bf* __restrict__ wbf, const float4* __restrict__ W, int n4)
{
    const int stride = gridDim.x * blockDim.x;
    for (int i = blockIdx.x * blockDim.x + threadIdx.x; i < n4; i += stride) {
        const float4 v = W[i];
        v4bf o = { (__bf16)v.x, (__bf16)v.y, (__bf16)v.z, (__bf16)v.w };
        wbf[i] = o;
    }
}

extern "C" void kernel_launch(void* const* d_in, const int* in_sizes, int n_in,
                              void* d_out, int out_size, void* d_ws, size_t ws_size,
                              hipStream_t stream)
{
    const float* x    = (const float*)d_in[0];   // [BATCH, NDIM]
    const float* Ws   = (const float*)d_in[1];   // [DEPTH, HDIM, NDIM]
    const float* bs   = (const float*)d_in[2];   // [DEPTH, HDIM]
    const int*   idxs = (const int*)  d_in[3];   // [DEPTH, HDIM]

    float* xw = (float*)d_out;                   // working x (f32), final output

    // Workspace layout: hbuf f32 [B,H] | xbf bf16 [B,N] | wbf bf16 [H,N]
    char* ws = (char*)d_ws;
    float*  hbuf = (float*)ws;
    __bf16* xbf  = (__bf16*)(ws + (size_t)BATCH * HDIM * 4);
    __bf16* wbf  = (__bf16*)(ws + (size_t)BATCH * HDIM * 4 + (size_t)BATCH * NDIM * 2);

    const int nx4 = BATCH * NDIM / 4;
    mshuffle_init_x<<<dim3(4096), dim3(256), 0, stream>>>(
        (float4*)xw, (v4bf*)xbf, (const float4*)x, nx4);

    const dim3 ggrid(BATCH / BM, HDIM / BN);     // 16 x 16 = 256 blocks
    const int nw4 = HDIM * NDIM / 4;
    for (int d = 0; d < DEPTH; ++d) {
        mshuffle_convert_w<<<dim3(2048), dim3(256), 0, stream>>>(
            (v4bf*)wbf, (const float4*)(Ws + (size_t)d * HDIM * NDIM), nw4);
        mshuffle_gemm_softplus<<<ggrid, dim3(256), 0, stream>>>(
            xbf, wbf, bs + (size_t)d * HDIM, hbuf);
        mshuffle_scatter<<<dim3(2048), dim3(256), 0, stream>>>(
            xw, xbf, hbuf, idxs + (size_t)d * HDIM);
    }
}
